// Xerxes2MoeMLPStack_601295421790
// MI455X (gfx1250) — matrix operand order repaired and checked by reading, compiled, not run on hardware
//
#include <hip/hip_runtime.h>
#include <hip/hip_bf16.h>
#include <math.h>

// ---------------------------------------------------------------------------
// MoE MLP stack (grouped GEMM) for gfx1250:
//   h = silu(X @ Wg[e]) * (X @ Wu[e])   -> bf16 workspace
//   out = h @ Wd[e]                     -> f32
// bf16 WMMA (v_wmma_f32_16x16x32_bf16) with f32 accumulation.
// Software-pipelined staging (global prefetch regs -> LDS), clamped (branch-
// free) ragged-row handling, async LDS copy for the bf16 A tile in kernel 2.
// ---------------------------------------------------------------------------

typedef __bf16 bf16t;
typedef __attribute__((ext_vector_type(16))) __bf16 v16bf;
typedef __attribute__((ext_vector_type(8)))  __bf16 v8bf;
typedef __attribute__((ext_vector_type(4)))  __bf16 v4bf;
typedef __attribute__((ext_vector_type(8)))  float  v8f;
typedef __attribute__((ext_vector_type(4)))  int    v4i;

// Address-space qualified pointee types for the async-LDS builtin
// (signature revealed by clang: (AS1 v4i*, AS3 v4i*, imm, imm)).
typedef __attribute__((address_space(1))) v4i v4i_g;
typedef __attribute__((address_space(3))) v4i v4i_l;

#define EXPERTS 8
#define HDIM    2048
#define IDIM    1024
#define TM      128
#define TN      128
#define TK      32
#define LDSB    40   // LDS K-stride in halves: 80 bytes, 16B aligned

#if __has_builtin(__builtin_amdgcn_global_load_async_to_lds_b128)
#define HAVE_ASYNC_LDS 1
#else
#define HAVE_ASYNC_LDS 0
#endif

__device__ __forceinline__ void wait_asynccnt0() {
#if __has_builtin(__builtin_amdgcn_s_wait_asynccnt)
  __builtin_amdgcn_s_wait_asynccnt(0);
#else
  asm volatile("s_wait_asynccnt 0x0" ::: "memory");
#endif
}

// Map a linear M-tile index onto (expert, row range) given ragged group sizes.
__device__ __forceinline__ int map_tile(const int* __restrict__ gs, int tile,
                                        int* row0, int* row_end) {
  int off = 0;
  for (int e = 0; e < EXPERTS; ++e) {
    int g  = gs[e];
    int nt = (g + TM - 1) / TM;
    if (tile < nt) { *row0 = off + tile * TM; *row_end = off + g; return e; }
    tile -= nt; off += g;
  }
  return -1;
}

// Build a 16x32 (A) or 32x16 (B) bf16 WMMA fragment from two 16-byte LDS loads.
__device__ __forceinline__ v16bf make_frag(const bf16t* __restrict__ base,
                                           int o_lo, int o_hi) {
  v8bf lo = *(const v8bf*)(base + o_lo);
  v8bf hi = *(const v8bf*)(base + o_hi);
  return __builtin_shufflevector(lo, hi, 0, 1, 2, 3, 4, 5, 6, 7,
                                         8, 9, 10, 11, 12, 13, 14, 15);
}

__device__ __forceinline__ v4bf cvt4(float4 f) {
  v4bf h;
  h[0] = (__bf16)f.x; h[1] = (__bf16)f.y;
  h[2] = (__bf16)f.z; h[3] = (__bf16)f.w;
  return h;
}

// ---------------------------------------------------------------------------
// Kernel 1: fused gate+up GEMM + SwiGLU.  C tile 128x128 over [T, I].
// ---------------------------------------------------------------------------
extern "C" __global__ void __launch_bounds__(256)
moe_gateup(const float* __restrict__ X, const float* __restrict__ Wg,
           const float* __restrict__ Wu, const int* __restrict__ gs,
           bf16t* __restrict__ Hws) {
  __shared__ __align__(16) bf16t sA[TM * LDSB];
  __shared__ __align__(16) bf16t sG[TN * LDSB];
  __shared__ __align__(16) bf16t sU[TN * LDSB];

  int row0, row_end;
  const int e = map_tile(gs, (int)blockIdx.x, &row0, &row_end);
  if (e < 0) return;
  const int n0   = (int)blockIdx.y * TN;
  const int last = row_end - 1;

  const int tid    = threadIdx.x;
  const int lane   = tid & 31;
  const int wid    = tid >> 5;
  const int laneLo = lane & 15;
  const int laneHi = lane >> 4;
  const int wm     = wid & 3;   // 4 waves along M (32 rows each)
  const int wn     = wid >> 2;  // 2 waves along N (64 cols each)

  const float* wg = Wg + (size_t)e * HDIM * IDIM + n0;  // uniform bases
  const float* wu = Wu + (size_t)e * HDIM * IDIM + n0;

  // K-invariant staging geometry (branch-free via row clamping).
  int aSrc[4], aDst[4], bSrc[4], bDst[4];
#pragma unroll
  for (int j = 0; j < 4; ++j) {
    const int idx = tid + 256 * j;
    {  // A: 128 rows x 32 K, 8 float4 per row
      const int r  = idx >> 3;
      const int kv = (idx & 7) << 2;
      int grow = row0 + r; grow = grow < last ? grow : last;  // clamp
      aSrc[j] = grow * HDIM + kv;
      aDst[j] = r * LDSB + kv;
    }
    {  // B: 32 K x 128 N, 32 float4 per K-row
      const int kk = idx >> 5;
      const int nv = (idx & 31) << 2;
      bSrc[j] = kk * IDIM + nv;
      bDst[j] = nv * LDSB + kk;   // +t*LDSB for component t
    }
  }

  const v8f vzero = {0.f, 0.f, 0.f, 0.f, 0.f, 0.f, 0.f, 0.f};
  v8f cg[2][4], cu[2][4];
#pragma unroll
  for (int i = 0; i < 2; ++i)
#pragma unroll
    for (int j = 0; j < 4; ++j) { cg[i][j] = vzero; cu[i][j] = vzero; }

  // ---- prefetch K-step 0 into registers ----
  float4 pa[4], pg[4], pu[4];
#pragma unroll
  for (int j = 0; j < 4; ++j) {
    pa[j] = *(const float4*)(X  + aSrc[j]);
    pg[j] = *(const float4*)(wg + bSrc[j]);
    pu[j] = *(const float4*)(wu + bSrc[j]);
  }

  for (int k0 = 0; k0 < HDIM; k0 += TK) {
    __syncthreads();               // previous compute finished reading LDS
    // ---- drain prefetch regs -> LDS (f32 -> bf16) ----
#pragma unroll
    for (int j = 0; j < 4; ++j)
      *(v4bf*)(sA + aDst[j]) = cvt4(pa[j]);
#pragma unroll
    for (int j = 0; j < 4; ++j) {
      sG[bDst[j] + 0 * LDSB] = (__bf16)pg[j].x;
      sG[bDst[j] + 1 * LDSB] = (__bf16)pg[j].y;
      sG[bDst[j] + 2 * LDSB] = (__bf16)pg[j].z;
      sG[bDst[j] + 3 * LDSB] = (__bf16)pg[j].w;
      sU[bDst[j] + 0 * LDSB] = (__bf16)pu[j].x;
      sU[bDst[j] + 1 * LDSB] = (__bf16)pu[j].y;
      sU[bDst[j] + 2 * LDSB] = (__bf16)pu[j].z;
      sU[bDst[j] + 3 * LDSB] = (__bf16)pu[j].w;
    }
    __syncthreads();
    // ---- issue next K-step's global loads (overlap with WMMA) ----
    if (k0 + TK < HDIM) {
      const int ka = k0 + TK;
      const int kb = (k0 + TK) * IDIM;
#pragma unroll
      for (int j = 0; j < 4; ++j) {
        pa[j] = *(const float4*)(X  + aSrc[j] + ka);
        pg[j] = *(const float4*)(wg + bSrc[j] + kb);
        pu[j] = *(const float4*)(wu + bSrc[j] + kb);
      }
    }

    // ---- fragments + WMMA ----
    v16bf a[2], bg[4], bu[4];
#pragma unroll
    for (int mi = 0; mi < 2; ++mi) {
      const int arow = wm * 32 + mi * 16 + laneLo;
      const int o = arow * LDSB + laneHi * 8;
      a[mi] = make_frag(sA, o, o + 16);
    }
#pragma unroll
    for (int ni = 0; ni < 4; ++ni) {
      const int bn = wn * 64 + ni * 16 + laneLo;
      const int o = bn * LDSB + laneHi * 8;
      bg[ni] = make_frag(sG, o, o + 16);
      bu[ni] = make_frag(sU, o, o + 16);
    }
#pragma unroll
    for (int mi = 0; mi < 2; ++mi)
#pragma unroll
      for (int ni = 0; ni < 4; ++ni) {
        cg[mi][ni] = __builtin_amdgcn_wmma_f32_16x16x32_bf16(
            false, a[mi], false, bg[ni], (short)0, cg[mi][ni], false, false);
        cu[mi][ni] = __builtin_amdgcn_wmma_f32_16x16x32_bf16(
            false, a[mi], false, bu[ni], (short)0, cu[mi][ni], false, false);
      }
  }

  // ---- epilogue: SwiGLU, write bf16 h ----
#pragma unroll
  for (int mi = 0; mi < 2; ++mi)
#pragma unroll
    for (int ni = 0; ni < 4; ++ni) {
      const int col = n0 + wn * 64 + ni * 16 + laneLo;
#pragma unroll
      for (int r = 0; r < 8; ++r) {
        const int row = row0 + wm * 32 + mi * 16 + laneHi * 8 + r;
        if (row < row_end) {
          const float g = cg[mi][ni][r];
          const float u = cu[mi][ni][r];
          const float h = (g / (1.0f + __expf(-g))) * u;
          Hws[(size_t)row * IDIM + col] = (__bf16)h;
        }
      }
    }
}

// ---------------------------------------------------------------------------
// Kernel 2: down GEMM.  C tile 128x128 over [T, H], f32 output.
// A tile (bf16) is copied to LDS with async LDS loads when available.
// ---------------------------------------------------------------------------
extern "C" __global__ void __launch_bounds__(256)
moe_down(const bf16t* __restrict__ Hws, const float* __restrict__ Wd,
         const int* __restrict__ gs, float* __restrict__ Out) {
  __shared__ __align__(16) bf16t sA[TM * LDSB];
  __shared__ __align__(16) bf16t sB[TN * LDSB];

  int row0, row_end;
  const int e = map_tile(gs, (int)blockIdx.x, &row0, &row_end);
  if (e < 0) return;
  const int n0   = (int)blockIdx.y * TN;
  const int last = row_end - 1;

  const int tid    = threadIdx.x;
  const int lane   = tid & 31;
  const int wid    = tid >> 5;
  const int laneLo = lane & 15;
  const int laneHi = lane >> 4;
  const int wm     = wid & 3;
  const int wn     = wid >> 2;

  const float* wd = Wd + (size_t)e * IDIM * HDIM + n0;

  int aSrc[2], aDst[2], bSrc[4], bDst[4];
#pragma unroll
  for (int j = 0; j < 2; ++j) {  // A: 128 rows x 32 halves, 4 uint4 per row
    const int idx = tid + 256 * j;
    const int r   = idx >> 2;
    const int kv  = (idx & 3) << 3;  // half offset
    int grow = row0 + r; grow = grow < last ? grow : last;
    aSrc[j] = grow * IDIM + kv;
    aDst[j] = r * LDSB + kv;
  }
#pragma unroll
  for (int j = 0; j < 4; ++j) {  // B: 32 K x 128 N f32
    const int idx = tid + 256 * j;
    const int kk  = idx >> 5;
    const int nv  = (idx & 31) << 2;
    bSrc[j] = kk * HDIM + nv;
    bDst[j] = nv * LDSB + kk;
  }

  const v8f vzero = {0.f, 0.f, 0.f, 0.f, 0.f, 0.f, 0.f, 0.f};
  v8f c[2][4];
#pragma unroll
  for (int i = 0; i < 2; ++i)
#pragma unroll
    for (int j = 0; j < 4; ++j) c[i][j] = vzero;

  // ---- prefetch K-step 0 ----
  float4 pb[4];
#if !HAVE_ASYNC_LDS
  uint4 pA[2];
#pragma unroll
  for (int j = 0; j < 2; ++j) pA[j] = *(const uint4*)(Hws + aSrc[j]);
#endif
#pragma unroll
  for (int j = 0; j < 4; ++j) pb[j] = *(const float4*)(wd + bSrc[j]);

  for (int k0 = 0; k0 < IDIM; k0 += TK) {
    __syncthreads();
    // ---- A tile: bf16 copy global -> LDS ----
#if HAVE_ASYNC_LDS
#pragma unroll
    for (int j = 0; j < 2; ++j)
      __builtin_amdgcn_global_load_async_to_lds_b128(
          (v4i_g*)(bf16t*)(Hws + aSrc[j] + k0),
          (v4i_l*)(sA + aDst[j]),
          0, 0);
#else
#pragma unroll
    for (int j = 0; j < 2; ++j) *(uint4*)(sA + aDst[j]) = pA[j];
#endif
    // ---- B tile: f32 -> bf16, transposed [N][K] ----
#pragma unroll
    for (int j = 0; j < 4; ++j) {
      sB[bDst[j] + 0 * LDSB] = (__bf16)pb[j].x;
      sB[bDst[j] + 1 * LDSB] = (__bf16)pb[j].y;
      sB[bDst[j] + 2 * LDSB] = (__bf16)pb[j].z;
      sB[bDst[j] + 3 * LDSB] = (__bf16)pb[j].w;
    }
#if HAVE_ASYNC_LDS
    wait_asynccnt0();
#endif
    __syncthreads();
    // ---- issue next K-step's global loads ----
    if (k0 + TK < IDIM) {
      const int kb = (k0 + TK) * HDIM;
#pragma unroll
      for (int j = 0; j < 4; ++j) pb[j] = *(const float4*)(wd + bSrc[j] + kb);
#if !HAVE_ASYNC_LDS
#pragma unroll
      for (int j = 0; j < 2; ++j)
        pA[j] = *(const uint4*)(Hws + aSrc[j] + (k0 + TK));
#endif
    }

    v16bf a[2], b[4];
#pragma unroll
    for (int mi = 0; mi < 2; ++mi) {
      const int arow = wm * 32 + mi * 16 + laneLo;
      const int o = arow * LDSB + laneHi * 8;
      a[mi] = make_frag(sA, o, o + 16);
    }
#pragma unroll
    for (int ni = 0; ni < 4; ++ni) {
      const int bn = wn * 64 + ni * 16 + laneLo;
      const int o = bn * LDSB + laneHi * 8;
      b[ni] = make_frag(sB, o, o + 16);
    }
#pragma unroll
    for (int mi = 0; mi < 2; ++mi)
#pragma unroll
      for (int ni = 0; ni < 4; ++ni)
        c[mi][ni] = __builtin_amdgcn_wmma_f32_16x16x32_bf16(
            false, a[mi], false, b[ni], (short)0, c[mi][ni], false, false);
  }

  // ---- epilogue: f32 output ----
#pragma unroll
  for (int mi = 0; mi < 2; ++mi)
#pragma unroll
    for (int ni = 0; ni < 4; ++ni) {
      const int col = n0 + wn * 64 + ni * 16 + laneLo;
#pragma unroll
      for (int r = 0; r < 8; ++r) {
        const int row = row0 + wm * 32 + mi * 16 + laneHi * 8 + r;
        if (row < row_end)
          Out[(size_t)row * HDIM + col] = c[mi][ni][r];
      }
    }
}

// ---------------------------------------------------------------------------
extern "C" void kernel_launch(void* const* d_in, const int* in_sizes, int n_in,
                              void* d_out, int out_size, void* d_ws, size_t ws_size,
                              hipStream_t stream) {
  const float* X  = (const float*)d_in[0];   // [T, H] f32
  const float* Wg = (const float*)d_in[1];   // [E, H, I] f32
  const float* Wu = (const float*)d_in[2];   // [E, H, I] f32
  const float* Wd = (const float*)d_in[3];   // [E, I, H] f32
  const int*   gs = (const int*)d_in[4];     // [E] i32
  float* out = (float*)d_out;                // [T, H] f32
  bf16t* Hws = (bf16t*)d_ws;                 // [T, I] bf16 intermediate

  const int T = in_sizes[0] / HDIM;
  const int mtiles = (T + TM - 1) / TM + EXPERTS;  // upper bound incl. ragged padding

  dim3 blk(256);
  moe_gateup<<<dim3(mtiles, IDIM / TN), blk, 0, stream>>>(X, Wg, Wu, gs, Hws);
  moe_down  <<<dim3(mtiles, HDIM / TN), blk, 0, stream>>>(Hws, Wd, gs, out);
}
